// CausalSelfAttention_70935679861415
// MI455X (gfx1250) — matrix operand order repaired and checked by reading, not compiled, not run on hardware
//
#include <hip/hip_runtime.h>
#include <hip/hip_bf16.h>

// Problem constants (match reference)
#define T_SEQ  2048
#define C_DIM  1024
#define H_NUM  16
#define D_HEAD 64
#define B_NUM  4
#define M_ROWS (B_NUM * T_SEQ)   // 8192

typedef __attribute__((ext_vector_type(16))) __bf16 v16bf;
typedef __attribute__((ext_vector_type(8)))  __bf16 bf16x8;
typedef __attribute__((ext_vector_type(8)))  float  v8f;
typedef __attribute__((ext_vector_type(4)))  unsigned int u32x4;
typedef __attribute__((ext_vector_type(8)))  int i32x8;
typedef __attribute__((ext_vector_type(4)))  int i32x4;

union FragBF { v16bf v; bf16x8 h[2]; i32x4 d[2]; };

// Native bf16 convert (gfx1250 has hardware BF16 conversion; RNE)
static __device__ __forceinline__ __bf16 f2bf(float f) { return (__bf16)f; }

static __device__ __forceinline__ v8f wmma_bf16(v16bf a, v16bf b, v8f c) {
  return __builtin_amdgcn_wmma_f32_16x16x32_bf16(false, a, false, b, (short)0, c,
                                                 false, false);
}

// ---- Fragment loaders (wave32 WMMA 16-bit operands; this file's contiguous k order, B matches) ----
// A 16x32: lane l holds row m=l&15; K = {kb..kb+15}, kb=(l<16)?0:16
static __device__ __forceinline__ v16bf load_a_frag(const __bf16* A, int lda, int lane) {
  FragBF f;
  int r  = lane & 15;
  int kb = (lane < 16) ? 0 : 16;
  const __bf16* p = A + (size_t)r * lda + kb;
  f.h[0] = *(const bf16x8*)(p);
  f.h[1] = *(const bf16x8*)(p + 8);
  return f.v;
}

// ---- TDM: DMA a 64(n) x 32(k) bf16 tile of WT[1024][1024] into LDS ----
// D# per CDNA5 ISA ch.8: group0 = {flags, lds_addr, global_addr|type},
// group1 = dims/strides (data_size=2B, tile 32x64, tensor 1024x1024, stride 1024).
static __device__ __forceinline__ void tdm_load_wtile(const __bf16* gsrc,
                                                      unsigned lds_byte_addr) {
  unsigned long long ga = (unsigned long long)(const void*)gsrc;
  u32x4 g0;
  g0[0] = 0x1u;                                            // count=1, user mode
  g0[1] = lds_byte_addr;                                   // lds_addr
  g0[2] = (unsigned)(ga & 0xFFFFFFFFu);                    // global_addr[31:0]
  g0[3] = (unsigned)((ga >> 32) & 0x01FFFFFFu) | (2u << 30); // addr[56:32] | type=2
  i32x8 g1;
  g1[0] = 0x00010000;          // workgroup_mask=0, data_size=1 (2 bytes)
  g1[1] = (int)(1024u << 16);  // tensor_dim0[15:0]=1024
  g1[2] = (int)(1024u << 16);  // tensor_dim0[31:16]=0 | tensor_dim1[15:0]=1024
  g1[3] = (int)(32u << 16);    // tensor_dim1[31:16]=0 | tile_dim0=32 (k)
  g1[4] = 64;                  // tile_dim1=64 (n), tile_dim2=0
  g1[5] = 1024;                // tensor_dim0_stride[31:0]=1024
  g1[6] = 0;                   // stride hi | tensor_dim1_stride lo
  g1[7] = 0;
  i32x4 z4 = (i32x4)0;
#if __has_include(<hip/amd_detail/amd_gfx1250_TDM.h>)
  i32x8 z8 = (i32x8)0;
  __builtin_amdgcn_tensor_load_to_lds(g0, g1, z4, z4, z8, 0);
#else
  __builtin_amdgcn_tensor_load_to_lds(g0, g1, z4, z4, 0);
#endif
}

// ---- Load the four 32x16 B-fragments of one 64x32 LDS tile via inline asm ----
// The tile is written by the TDM, which the compiler cannot model as stores to
// bsm; plain C loads get folded to undef ("no store to this LDS object").
// Inline ds_load_b128 + s_wait_dscnt keeps the reads real. addr = LDS byte
// address of this lane's first 16B: tile_base + n*64 + half*32.
static __device__ __forceinline__ void lds_load_bfrags(unsigned addr, FragBF bf[4]) {
  asm volatile(
      "ds_load_b128 %0, %8 offset:0\n\t"
      "ds_load_b128 %1, %8 offset:16\n\t"
      "ds_load_b128 %2, %8 offset:1024\n\t"
      "ds_load_b128 %3, %8 offset:1040\n\t"
      "ds_load_b128 %4, %8 offset:2048\n\t"
      "ds_load_b128 %5, %8 offset:2064\n\t"
      "ds_load_b128 %6, %8 offset:3072\n\t"
      "ds_load_b128 %7, %8 offset:3088\n\t"
      "s_wait_dscnt 0x0"
      : "=&v"(bf[0].d[0]), "=&v"(bf[0].d[1]),
        "=&v"(bf[1].d[0]), "=&v"(bf[1].d[1]),
        "=&v"(bf[2].d[0]), "=&v"(bf[2].d[1]),
        "=&v"(bf[3].d[0]), "=&v"(bf[3].d[1])
      : "v"(addr)
      : "memory");
}

// ---- GEMM core: wave computes 32(M) x 64(N); B tile staged in LDS by TDM ----
// bsm: __shared__ __bf16[2][64*32]; double buffered; last k-iteration peeled.
static __device__ __forceinline__ void gemm_wave_32x64_lds(
    const __bf16* __restrict__ Arow,   // &A[mbase * C_DIM] (32 rows)
    const __bf16* __restrict__ WT,     // [N][C] bf16 (transposed weights)
    int nbase, int lane, int wave, __bf16* bsm,
    v8f acc0[4], v8f acc1[4]) {
  const int half = lane >> 4;
  const int n    = lane & 15;
  const __bf16* Arow1 = Arow + 16 * C_DIM;
  // LDS byte addresses: low 32 bits of a generic LDS pointer = LDS address.
  const unsigned lds_base = (unsigned)(size_t)(void*)bsm;
  const unsigned lane_off = (unsigned)(n * 64 + half * 32);

  if (wave == 0) tdm_load_wtile(WT + (size_t)nbase * C_DIM, lds_base);
  v16bf af0 = load_a_frag(Arow,  C_DIM, lane);
  v16bf af1 = load_a_frag(Arow1, C_DIM, lane);

  #pragma unroll 1
  for (int kk = 0; kk < 31; ++kk) {
    if (wave == 0) __builtin_amdgcn_s_wait_tensorcnt((short)0);
    __syncthreads();                   // buf[cur] ready; buf[nxt] reads done
    if (wave == 0)
      tdm_load_wtile(WT + (size_t)nbase * C_DIM + (kk + 1) * 32,
                     lds_base + (unsigned)(((kk + 1) & 1) * 4096));
    // prefetch next A fragments while computing on current ones
    v16bf a0n = load_a_frag(Arow  + (kk + 1) * 32, C_DIM, lane);
    v16bf a1n = load_a_frag(Arow1 + (kk + 1) * 32, C_DIM, lane);

    FragBF bfr[4];
    lds_load_bfrags(lds_base + (unsigned)((kk & 1) * 4096) + lane_off, bfr);
    #pragma unroll
    for (int j = 0; j < 4; ++j) {
      acc0[j] = wmma_bf16(af0, bfr[j].v, acc0[j]);
      acc1[j] = wmma_bf16(af1, bfr[j].v, acc1[j]);
    }
    af0 = a0n;
    af1 = a1n;
  }
  // peeled final iteration (kk == 31, buffer 1)
  if (wave == 0) __builtin_amdgcn_s_wait_tensorcnt((short)0);
  __syncthreads();
  {
    FragBF bfr[4];
    lds_load_bfrags(lds_base + 4096u + lane_off, bfr);
    #pragma unroll
    for (int j = 0; j < 4; ++j) {
      acc0[j] = wmma_bf16(af0, bfr[j].v, acc0[j]);
      acc1[j] = wmma_bf16(af1, bfr[j].v, acc1[j]);
    }
  }
}

// ---- Kernel 1a: fp32 -> bf16 conversion of x ----
__global__ void __launch_bounds__(256) convert_x_kernel(
    const float* __restrict__ x, __bf16* __restrict__ xb, int nElem) {
  int i = blockIdx.x * blockDim.x + threadIdx.x;
  int stride = gridDim.x * blockDim.x;
  for (; i < nElem; i += stride) xb[i] = f2bf(x[i]);
}

// ---- Kernel 1b: W[C][C] fp32 -> WT[C][C] bf16 (WT[n][k] = W[k][n]) ----
__global__ void __launch_bounds__(256) transpose_w_kernel(
    const float* __restrict__ W, __bf16* __restrict__ WT) {
  int i = blockIdx.x * blockDim.x + threadIdx.x;   // 1M threads
  int n = i >> 10;
  int k = i & (C_DIM - 1);
  WT[i] = f2bf(W[(size_t)k * C_DIM + n]);
}

// ---- Kernel 2: fused QKV projection. blockIdx.z selects Q/K/V. ----
__global__ void __launch_bounds__(256) qkv_gemm_kernel(
    const __bf16* __restrict__ xb,
    const __bf16* __restrict__ WTq, const __bf16* __restrict__ WTk,
    const __bf16* __restrict__ WTv,
    const float* __restrict__ bq, const float* __restrict__ bk,
    const float* __restrict__ bv,
    __bf16* __restrict__ Qo, __bf16* __restrict__ Ko, __bf16* __restrict__ VTo) {
  __shared__ __bf16 bsm[2 * 64 * 32];   // double-buffered TDM B tile
  int lane = threadIdx.x & 31, wave = threadIdx.x >> 5;
  int mbase = blockIdx.x * 256 + wave * 32;
  int nbase = blockIdx.y * 64;
  int mode  = blockIdx.z;  // 0=Q 1=K 2=V
  const __bf16* WT  = (mode == 0) ? WTq : (mode == 1) ? WTk : WTv;
  const float* bias = (mode == 0) ? bq  : (mode == 1) ? bk  : bv;

  v8f acc0[4], acc1[4];
  #pragma unroll
  for (int j = 0; j < 4; ++j) { acc0[j] = (v8f)0.0f; acc1[j] = (v8f)0.0f; }

  gemm_wave_32x64_lds(xb + (size_t)mbase * C_DIM, WT, nbase, lane, wave, bsm,
                      acc0, acc1);

  int half = lane >> 4, n = lane & 15;
  #pragma unroll
  for (int mt = 0; mt < 2; ++mt) {
    v8f* acc = mt ? acc1 : acc0;
    int mb = mbase + mt * 16;
    #pragma unroll
    for (int j = 0; j < 4; ++j) {
      int ng = nbase + j * 16 + n;
      int h = ng >> 6, d = ng & (D_HEAD - 1);
      float bb = bias[ng];
      #pragma unroll
      for (int v = 0; v < 8; ++v) {
        int mg = mb + v + 8 * half;           // global row = b*T + t
        int b = mg >> 11, t = mg & (T_SEQ - 1);
        float val = acc[j][v] + bb;
        if (mode == 2)
          VTo[(((size_t)(b * H_NUM + h)) * D_HEAD + d) * T_SEQ + t] = f2bf(val);
        else {
          __bf16* dst = (mode == 0) ? Qo : Ko;
          dst[(((size_t)(b * H_NUM + h)) * T_SEQ + t) * D_HEAD + d] = f2bf(val);
        }
      }
    }
  }
}

// ---- Kernel 3: causal flash attention, one wave per 16-row query tile ----
__global__ void __launch_bounds__(256) attn_kernel(
    const __bf16* __restrict__ Q, const __bf16* __restrict__ K,
    const __bf16* __restrict__ VT, __bf16* __restrict__ Yb) {
  __shared__ __bf16 sm[8 * 16 * 32];   // 1KB / wave P-staging (D-layout -> A-layout)
  int lane = threadIdx.x & 31, wave = threadIdx.x >> 5;
  int bh = blockIdx.y;
  int b = bh >> 4, h = bh & (H_NUM - 1);
  const __bf16* Qh = Q  + (size_t)bh * T_SEQ * D_HEAD;
  const __bf16* Kh = K  + (size_t)bh * T_SEQ * D_HEAD;
  const __bf16* Vh = VT + (size_t)bh * D_HEAD * T_SEQ;
  __bf16* smw = sm + wave * (16 * 32);
  int qbase = blockIdx.x * 128 + wave * 16;

  int r = lane & 15;
  int kbA = (lane < 16) ? 0 : 16;
  int half = lane >> 4, n = lane & 15;
  int kbB = (lane < 16) ? 0 : 16;

  // Q tile fragments: two 16x32 A-frags covering D=0..31 and D=32..63
  FragBF q0, q1;
  {
    const __bf16* qp = Qh + (size_t)(qbase + r) * D_HEAD + kbA;
    q0.h[0] = *(const bf16x8*)(qp);        q0.h[1] = *(const bf16x8*)(qp + 8);
    q1.h[0] = *(const bf16x8*)(qp + 32);   q1.h[1] = *(const bf16x8*)(qp + 40);
  }

  float mrow[8], lrow[8];
  v8f acc[4];
  #pragma unroll
  for (int j = 0; j < 4; ++j) acc[j] = (v8f)0.0f;
  #pragma unroll
  for (int v = 0; v < 8; ++v) { mrow[v] = -3.0e38f; lrow[v] = 0.0f; }

  const float scale = 0.125f;   // 1/sqrt(64)

  for (int s0 = 0; s0 < qbase + 16; s0 += 32) {
    // prefetch next key/value block into cache (global_prefetch_b8)
    if (s0 + 32 < qbase + 16) {
      __builtin_prefetch(Kh + (size_t)(s0 + 32 + n) * D_HEAD, 0, 1);
      __builtin_prefetch(Kh + (size_t)(s0 + 48 + n) * D_HEAD, 0, 1);
      __builtin_prefetch(Vh + (size_t)(n) * T_SEQ + s0 + 32, 0, 1);
      __builtin_prefetch(Vh + (size_t)(48 + n) * T_SEQ + s0 + 32, 0, 1);
    }

    // ---- load all four K B-frags, then issue the 4 score WMMAs ----
    FragBF k00, k01, k10, k11;
    {
      const __bf16* kp0 = Kh + (size_t)(s0 + n) * D_HEAD + kbB;
      const __bf16* kp1 = Kh + (size_t)(s0 + 16 + n) * D_HEAD + kbB;
      k00.h[0] = *(const bf16x8*)(kp0);       k00.h[1] = *(const bf16x8*)(kp0 + 8);
      k01.h[0] = *(const bf16x8*)(kp0 + 32);  k01.h[1] = *(const bf16x8*)(kp0 + 40);
      k10.h[0] = *(const bf16x8*)(kp1);       k10.h[1] = *(const bf16x8*)(kp1 + 8);
      k11.h[0] = *(const bf16x8*)(kp1 + 32);  k11.h[1] = *(const bf16x8*)(kp1 + 40);
    }
    v8f sc0 = (v8f)0.0f, sc1 = (v8f)0.0f;
    sc0 = wmma_bf16(q0.v, k00.v, sc0);
    sc0 = wmma_bf16(q1.v, k01.v, sc0);
    sc1 = wmma_bf16(q0.v, k10.v, sc1);
    sc1 = wmma_bf16(q1.v, k11.v, sc1);

    // ---- hoist V B-frag loads above the softmax so latency hides there ----
    FragBF vf[4];
    #pragma unroll
    for (int j = 0; j < 4; ++j) {
      const __bf16* vp = Vh + (size_t)(j * 16 + n) * T_SEQ + s0 + kbB;
      vf[j].h[0] = *(const bf16x8*)(vp);
      vf[j].h[1] = *(const bf16x8*)(vp + 8);
    }

    // ---- causal mask + scale ----
    #pragma unroll
    for (int v = 0; v < 8; ++v) {
      int qg = qbase + v + 8 * half;
      float x0 = sc0[v] * scale;
      float x1 = sc1[v] * scale;
      sc0[v] = ((s0 + n)      <= qg) ? x0 : -3.0e38f;
      sc1[v] = ((s0 + 16 + n) <= qg) ? x1 : -3.0e38f;
    }

    // ---- online softmax (row reductions across 16-lane halves) ----
    #pragma unroll
    for (int v = 0; v < 8; ++v) {
      float rm = fmaxf(sc0[v], sc1[v]);
      #pragma unroll
      for (int msk = 1; msk < 16; msk <<= 1) rm = fmaxf(rm, __shfl_xor(rm, msk, 32));
      float nm  = fmaxf(mrow[v], rm);
      float fac = __expf(mrow[v] - nm);
      mrow[v] = nm;
      float p0 = __expf(sc0[v] - nm);
      float p1 = __expf(sc1[v] - nm);
      float rs = p0 + p1;
      #pragma unroll
      for (int msk = 1; msk < 16; msk <<= 1) rs += __shfl_xor(rs, msk, 32);
      lrow[v] = lrow[v] * fac + rs;
      #pragma unroll
      for (int j = 0; j < 4; ++j) acc[j][v] *= fac;
      int m2 = v + 8 * half;
      smw[m2 * 32 + n]      = f2bf(p0);   // P tile row-major in LDS
      smw[m2 * 32 + 16 + n] = f2bf(p1);
    }

    // ---- reload P as A-frag (transpose through LDS; same-wave DS is in-order) ----
    FragBF pf;
    {
      const __bf16* pp = smw + r * 32 + kbA;
      pf.h[0] = *(const bf16x8*)(pp);
      pf.h[1] = *(const bf16x8*)(pp + 8);
    }

    // ---- O += P * V ----
    #pragma unroll
    for (int j = 0; j < 4; ++j)
      acc[j] = wmma_bf16(pf.v, vf[j].v, acc[j]);
  }

  // ---- normalize and write Y [B,T,C] (merged heads) as bf16 ----
  #pragma unroll
  for (int j = 0; j < 4; ++j) {
    #pragma unroll
    for (int v = 0; v < 8; ++v) {
      int m2 = v + 8 * half;
      float o = acc[j][v] / lrow[v];
      Yb[((size_t)(b * T_SEQ) + qbase + m2) * C_DIM + h * D_HEAD + j * 16 + n] = f2bf(o);
    }
  }
}

// ---- Kernel 4: output projection, fp32 result ----
__global__ void __launch_bounds__(256) proj_gemm_kernel(
    const __bf16* __restrict__ yb, const __bf16* __restrict__ WTp,
    const float* __restrict__ bp, float* __restrict__ out) {
  __shared__ __bf16 bsm[2 * 64 * 32];   // double-buffered TDM B tile
  int lane = threadIdx.x & 31, wave = threadIdx.x >> 5;
  int mbase = blockIdx.x * 256 + wave * 32;
  int nbase = blockIdx.y * 64;

  v8f acc0[4], acc1[4];
  #pragma unroll
  for (int j = 0; j < 4; ++j) { acc0[j] = (v8f)0.0f; acc1[j] = (v8f)0.0f; }

  gemm_wave_32x64_lds(yb + (size_t)mbase * C_DIM, WTp, nbase, lane, wave, bsm,
                      acc0, acc1);

  int half = lane >> 4, n = lane & 15;
  #pragma unroll
  for (int mt = 0; mt < 2; ++mt) {
    v8f* acc = mt ? acc1 : acc0;
    int mb = mbase + mt * 16;
    #pragma unroll
    for (int j = 0; j < 4; ++j) {
      int ng = nbase + j * 16 + n;
      float bb = bp[ng];
      #pragma unroll
      for (int v = 0; v < 8; ++v) {
        int mg = mb + v + 8 * half;
        out[(size_t)mg * C_DIM + ng] = acc[j][v] + bb;
      }
    }
  }
}

extern "C" void kernel_launch(void* const* d_in, const int* in_sizes, int n_in,
                              void* d_out, int out_size, void* d_ws, size_t ws_size,
                              hipStream_t stream) {
  const float* x  = (const float*)d_in[0];
  // d_in[1] = causal mask, handled analytically
  const float* Wq = (const float*)d_in[2];
  const float* bq = (const float*)d_in[3];
  const float* Wk = (const float*)d_in[4];
  const float* bk = (const float*)d_in[5];
  const float* Wv = (const float*)d_in[6];
  const float* bv = (const float*)d_in[7];
  const float* Wp = (const float*)d_in[8];
  const float* bp = (const float*)d_in[9];

  char* w = (char*)d_ws;
  size_t off = 0;
  __bf16* xb  = (__bf16*)(w + off); off += (size_t)M_ROWS * C_DIM * 2;  // 16 MB
  __bf16* WTq = (__bf16*)(w + off); off += (size_t)C_DIM * C_DIM * 2;   //  2 MB
  __bf16* WTk = (__bf16*)(w + off); off += (size_t)C_DIM * C_DIM * 2;
  __bf16* WTv = (__bf16*)(w + off); off += (size_t)C_DIM * C_DIM * 2;
  __bf16* WTp = (__bf16*)(w + off); off += (size_t)C_DIM * C_DIM * 2;
  __bf16* Qb  = (__bf16*)(w + off); off += (size_t)M_ROWS * C_DIM * 2;  // [B,H,T,D]
  __bf16* Kb  = (__bf16*)(w + off); off += (size_t)M_ROWS * C_DIM * 2;  // [B,H,T,D]
  __bf16* VTb = (__bf16*)(w + off); off += (size_t)M_ROWS * C_DIM * 2;  // [B,H,D,T]
  __bf16* Yb  = (__bf16*)(w + off); off += (size_t)M_ROWS * C_DIM * 2;  // [B,T,C]

  convert_x_kernel<<<4096, 256, 0, stream>>>(x, xb, M_ROWS * C_DIM);
  transpose_w_kernel<<<4096, 256, 0, stream>>>(Wq, WTq);
  transpose_w_kernel<<<4096, 256, 0, stream>>>(Wk, WTk);
  transpose_w_kernel<<<4096, 256, 0, stream>>>(Wv, WTv);
  transpose_w_kernel<<<4096, 256, 0, stream>>>(Wp, WTp);

  qkv_gemm_kernel<<<dim3(32, 16, 3), 256, 0, stream>>>(
      xb, WTq, WTk, WTv, bq, bk, bv, Qb, Kb, VTb);

  attn_kernel<<<dim3(T_SEQ / 128, B_NUM * H_NUM), 256, 0, stream>>>(Qb, Kb, VTb, Yb);

  proj_gemm_kernel<<<dim3(32, 16), 256, 0, stream>>>(Yb, WTp, bp, (float*)d_out);
}
